// HyenaOperator_64716567216399
// MI455X (gfx1250) — matrix operand order repaired and checked by reading, not compiled, not run on hardware
//
#include <hip/hip_runtime.h>
#include <hip/hip_bf16.h>
#include <stdint.h>

// ---------------------------------------------------------------------------
// Hyena operator for MI455X (gfx1250, wave32, WMMA).
// Conv = blocked Toeplitz matmul on bf16 WMMA 16x16x32 with full N packing:
// one WMMA covers 4 batches x 4 position-subtiles (all 16 columns live).
// ---------------------------------------------------------------------------

typedef __attribute__((ext_vector_type(16))) __bf16 v16bf;
typedef __attribute__((ext_vector_type(8)))  float  v8f;

#define D_MODEL 768
#define L_SEQ   8192
#define B_SZ    4
#define NPROJ   (3 * D_MODEL)   // 2304
#define BL      (B_SZ * L_SEQ)  // 32768

// conv LDS geometry (elements)
#define ZPAD   64                    // front zero pad per z row (t<0 taps)
#define ZROW   (L_SEQ + ZPAD)        // 8256
#define RFRONT 32                    // front zero pad on reversed filter
#define RLEN   (RFRONT + L_SEQ + 64) // 8288 (tail pad = causal mask)
#define Z_ELEMS (B_SZ * ZROW)        // 33024
#define CONV_LDS_BYTES ((Z_ELEMS + 2 * RLEN) * 2)  // 99200

static __device__ __forceinline__ float silu_f(float x) {
    return x * (1.0f / (1.0f + __expf(-x)));
}

// ---------------------------------------------------------------------------
// fp32 -> bf16 vectorized convert
// ---------------------------------------------------------------------------
__global__ void f32_to_bf16_v4(const float* __restrict__ in,
                               __bf16* __restrict__ out, int n4) {
    int i = blockIdx.x * blockDim.x + threadIdx.x;
    if (i >= n4) return;
    const float4 f = ((const float4*)in)[i];
    union { __bf16 h[4]; uint2 u; } p;
    p.h[0] = (__bf16)f.x; p.h[1] = (__bf16)f.y;
    p.h[2] = (__bf16)f.z; p.h[3] = (__bf16)f.w;
    ((uint2*)out)[i] = p.u;
}

// ---------------------------------------------------------------------------
// HyenaFilter MLP: 1 -> 64 -> 64 -> 2*768, per position l.
// Writes kernels REVERSED in L (hrev[o][d][p] = h[o][d][L-1-p]) as bf16.
// ---------------------------------------------------------------------------
__global__ __launch_bounds__(256) void hyena_filter_mlp(
    const float* __restrict__ w1, const float* __restrict__ b1,
    const float* __restrict__ w2, const float* __restrict__ b2,
    const float* __restrict__ w3, const float* __restrict__ b3,
    __bf16* __restrict__ hrev) {
    __shared__ float h1[64];
    __shared__ float h2[64];
    const int t = threadIdx.x;
    const int l = blockIdx.x;
    const float pos = (float)l * (1.0f / 8191.0f);  // linspace(0,1,8192)
    if (t < 64) h1[t] = silu_f(pos * w1[t] + b1[t]);
    __syncthreads();
    if (t < 64) {
        float s = b2[t];
        #pragma unroll 8
        for (int i = 0; i < 64; ++i) s += h1[i] * w2[i * 64 + t];
        h2[t] = silu_f(s);
    }
    __syncthreads();
    for (int j = t; j < 2 * D_MODEL; j += 256) {
        float s = b3[j];
        #pragma unroll 8
        for (int i = 0; i < 64; ++i) s += h2[i] * w3[i * (2 * D_MODEL) + j];
        const int o  = j / D_MODEL;
        const int dd = j - o * D_MODEL;
        hrev[((size_t)(o * D_MODEL + dd)) * L_SEQ + (size_t)(L_SEQ - 1 - l)] = (__bf16)s;
    }
}

// ---------------------------------------------------------------------------
// bf16 WMMA GEMM: C(M,*) = A(M,K) @ B(K,N) + bias.
// Workgroup = 256 thr (8 waves), tile 128(M) x 64(N), K staged 32-wide in LDS.
// mode 0: in_proj epilogue -> split v(fp32)/z1/z2(bf16) in (B,D,L).
// mode 1: out_proj epilogue -> fp32 row-major (BL, N) into d_out.
// ---------------------------------------------------------------------------
#define ASTR 34
#define BSTR 34

__global__ __launch_bounds__(256) void gemm_bf16_wmma(
    const __bf16* __restrict__ Ag, const __bf16* __restrict__ Bg,
    const float* __restrict__ bias, int K, int N, int mode,
    float* __restrict__ out_v, __bf16* __restrict__ out_z1,
    __bf16* __restrict__ out_z2, float* __restrict__ out_f) {
    __shared__ unsigned short As[128 * ASTR];
    __shared__ unsigned short Bt[64 * BSTR];   // stored transposed: [n][k]

    const int t      = threadIdx.x;
    const int lane   = t & 31;
    const int wv     = t >> 5;
    const int tile_m = blockIdx.y;
    const int tile_n = blockIdx.x;
    const int ncol0  = tile_n * 64;

    v8f acc[4] = {};

    const int mrow = (wv * 16 + (lane & 15)) * ASTR;   // even -> dword aligned
    const int hiAh = (lane & 16) ? 8 : 0;              // A frag u32 offset (K/2)
    const int hiB  = (lane & 16) ? 16 : 0;             // B frag K offset

    for (int kb = 0; kb < K; kb += 32) {
        __syncthreads();
        {   // stage A 128x32 (each thread: 16 contiguous bf16)
            const int ar = t >> 1;
            const int ac = (t & 1) << 4;
            const uint4* ap4 =
                (const uint4*)(Ag + (size_t)(tile_m * 128 + ar) * K + kb + ac);
            uint4 a0 = ap4[0];
            uint4 a1 = ap4[1];
            if (kb + 32 < K) __builtin_prefetch((const char*)ap4 + 64, 0, 0);
            uint32_t* as = (uint32_t*)&As[ar * ASTR + ac];
            as[0] = a0.x; as[1] = a0.y; as[2] = a0.z; as[3] = a0.w;
            as[4] = a1.x; as[5] = a1.y; as[6] = a1.z; as[7] = a1.w;

            // stage B 32x64 transposed (each thread: 8 contiguous bf16)
            const int bk  = t >> 3;
            const int nof = (t & 7) << 3;
            const uint4* bp4 =
                (const uint4*)(Bg + (size_t)(kb + bk) * N + ncol0 + nof);
            union { uint4 q; unsigned short s[8]; } ub;
            ub.q = bp4[0];
            if (kb + 32 < K)
                __builtin_prefetch((const char*)bp4 + (size_t)32 * N * 2, 0, 0);
            #pragma unroll
            for (int j = 0; j < 8; ++j) Bt[(nof + j) * BSTR + bk] = ub.s[j];
        }
        __syncthreads();

        union Frag { uint32_t u[8]; v16bf v; };
        Frag fa;
        const uint32_t* ap = (const uint32_t*)&As[mrow];
        #pragma unroll
        for (int vv = 0; vv < 8; ++vv)
            fa.u[vv] = ap[vv + hiAh];

        const uint32_t* bbase = (const uint32_t*)&Bt[(lane & 15) * BSTR + hiB];
        #pragma unroll
        for (int nt = 0; nt < 4; ++nt) {
            Frag fb;
            const uint32_t* bp = bbase + nt * (16 * BSTR / 2);
            #pragma unroll
            for (int vv = 0; vv < 8; ++vv) fb.u[vv] = bp[vv];
            acc[nt] = __builtin_amdgcn_wmma_f32_16x16x32_bf16(
                false, fa.v, false, fb.v, (short)0, acc[nt], false, false);
        }
    }

    const int mr = tile_m * 128 + wv * 16 + ((lane & 16) ? 8 : 0);
    #pragma unroll
    for (int nt = 0; nt < 4; ++nt) {
        const int j  = ncol0 + nt * 16 + (lane & 15);
        const float bv = bias[j];
        if (mode == 0) {
            const int s    = (j >= 2 * D_MODEL) ? 2 : ((j >= D_MODEL) ? 1 : 0);
            const int dcol = j - s * D_MODEL;
            #pragma unroll
            for (int r = 0; r < 8; ++r) {
                const int row = mr + r;
                const int b   = row >> 13;
                const int l   = row & (L_SEQ - 1);
                const float val = acc[nt][r] + bv;
                const size_t oi = ((size_t)(b * D_MODEL + dcol)) * L_SEQ + l;
                if (s == 0)      out_v[oi]  = val;
                else if (s == 1) out_z1[oi] = (__bf16)val;
                else             out_z2[oi] = (__bf16)val;
            }
        } else {
            #pragma unroll
            for (int r = 0; r < 8; ++r) {
                const int row = mr + r;
                out_f[(size_t)row * N + j] = acc[nt][r] + bv;
            }
        }
    }
}

// ---------------------------------------------------------------------------
// Causal long conv as Toeplitz-block WMMA with full N packing, fused gating:
//   yout[b,d,l] = (sum_{t<=l} h[d,l-t] * z[b,d,t]) * yprev[b,d,l]
//
// One channel d per blockIdx.x; each wave owns 64-position groups P0=64*i0.
//   A[m,k]       = h[(P0-jb) + m - k]      (Toeplitz slice of reversed filter)
//   B[k, 4g+b]   = z[b, jb + 16g + k]      -> column 4g+b = position P0+16g+m
// All 16 WMMA columns live (4 batches x 4 position subtiles).
// Reversed filter kept in LDS TWICE (copy shifted by one element) so every
// lane's Toeplitz pair-load is dword aligned -> ds_load_2addr_b32.
// z rows staged with CDNA5 async global->LDS (ASYNCcnt) bulk copies.
// ---------------------------------------------------------------------------
__global__ __launch_bounds__(256) void conv_gate_wmma(
    const __bf16* __restrict__ zin, const __bf16* __restrict__ hrev,
    const float* __restrict__ yprev, float* __restrict__ yout) {
    extern __shared__ char cvsmem[];
    __bf16* Zl = (__bf16*)cvsmem;                    // [4][ZROW], +64 front pad
    __bf16* R0 = Zl + Z_ELEMS;                       // [RLEN] reversed filter
    __bf16* R1 = R0 + RLEN;                          // [RLEN] shifted copy

    const int t    = threadIdx.x;
    const int lane = t & 31;
    const int wv   = t >> 5;
    const int d    = blockIdx.x;
    const int q    = blockIdx.y;

    // ---- stage z via async global->LDS copies (16B per lane per issue) ----
    #pragma unroll
    for (int c = 0; c < 16; ++c) {
        const int e  = (c * 256 + t) * 8;       // element index in [0, 32768)
        const int n  = e >> 13;
        const int tt = e & (L_SEQ - 1);
        const unsigned long long src = (unsigned long long)(uintptr_t)(
            zin + ((size_t)(n * D_MODEL + d)) * L_SEQ + tt);
        const uint32_t dst = (uint32_t)((n * ZROW + ZPAD + tt) * 2);
        asm volatile("global_load_async_to_lds_b128 %0, %1, off"
                     :: "v"(dst), "v"(src) : "memory");
    }
    // z front pads (t<0 taps)
    {
        const int r = t >> 6, cpad = t & 63;
        Zl[r * ZROW + cpad] = (__bf16)0.0f;
    }
    // ---- stage reversed filter R0 with front/tail zero pads ----
    const __bf16* hd = hrev + (size_t)d * L_SEQ;
    #pragma unroll
    for (int c = 0; c < 8; ++c) {
        const int p = (c * 256 + t) * 4;
        *(uint2*)&R0[RFRONT + p] = *(const uint2*)(hd + p);
    }
    if (t < RFRONT) R0[t] = (__bf16)0.0f;
    if (t < 64) R0[RFRONT + L_SEQ + t] = (__bf16)0.0f;
    __syncthreads();
    // ---- build shifted copy R1[p] = R0[p+1] ----
    for (int p = t; p < RLEN - 1; p += 256) R1[p] = R0[p + 1];
    if (t == 0) R1[RLEN - 1] = (__bf16)0.0f;
    // all waves' async copies + R1 complete before compute
    asm volatile("s_wait_asynccnt 0x0" ::: "memory");
    __syncthreads();

    const int m    = lane & 15;
    const int hiAh = (lane & 16) ? 8 : 0;      // A frag u32 offset (K/2)
    const int hiB  = (lane & 16) ? 16 : 0;     // B frag K offset
    const int nb   = lane & 15;                // column index
    const int cb   = nb & 3;                   // batch
    const int cg   = nb >> 2;                  // position subtile
    const int rhi  = (lane & 16) ? 8 : 0;

    for (int i0 = q * 8 + wv; i0 < L_SEQ / 64; i0 += 32) {
        v8f acc = {};
        const int P0 = i0 * 64;
        const uint32_t* zp0 =
            (const uint32_t*)&Zl[cb * ZROW + ZPAD + 16 * cg + hiB];
        for (int jb = -64; jb <= P0; jb += 32) {
            // A fragment: dword-aligned via parity-selected filter copy
            const int rb = RFRONT + (L_SEQ - 1) - (P0 - jb) - m;
            const __bf16* rsel = (rb & 1) ? (R1 + (rb - 1)) : (R0 + rb);
            const uint32_t* rp = (const uint32_t*)rsel;
            union Frag { uint32_t u[8]; v16bf v; };
            Frag fa;
            #pragma unroll
            for (int vv = 0; vv < 8; ++vv)
                fa.u[vv] = rp[vv + hiAh];
            // B fragment: all 16 columns live
            Frag fb;
            const uint32_t* zp = zp0 + (jb >> 1);
            #pragma unroll
            for (int vv = 0; vv < 8; ++vv) fb.u[vv] = zp[vv];
            acc = __builtin_amdgcn_wmma_f32_16x16x32_bf16(
                false, fa.v, false, fb.v, (short)0, acc, false, false);
        }
        // epilogue: lane column 4g+b -> positions P0+16g+rhi..+7, batch b
        const size_t base =
            ((size_t)(cb * D_MODEL + d)) * L_SEQ + P0 + 16 * cg + rhi;
        #pragma unroll
        for (int r = 0; r < 8; ++r)
            yout[base + r] = acc[r] * yprev[base + r];
    }
}

// ---------------------------------------------------------------------------
// (B,D,L) fp32 -> (B*L, D) bf16 for the out_proj A operand
// ---------------------------------------------------------------------------
__global__ void bdl_to_bld_bf16(const float* __restrict__ ysrc,
                                __bf16* __restrict__ yb) {
    const int idx = blockIdx.x * blockDim.x + threadIdx.x;
    const int bl  = idx / D_MODEL;
    const int dd  = idx - bl * D_MODEL;
    const int b   = bl >> 13;
    const int l   = bl & (L_SEQ - 1);
    yb[idx] = (__bf16)ysrc[((size_t)(b * D_MODEL + dd)) * L_SEQ + l];
}

// ---------------------------------------------------------------------------
extern "C" void kernel_launch(void* const* d_in, const int* in_sizes, int n_in,
                              void* d_out, int out_size, void* d_ws,
                              size_t ws_size, hipStream_t stream) {
    (void)in_sizes; (void)n_in; (void)out_size; (void)ws_size;
    const float* x     = (const float*)d_in[0];
    const float* in_w  = (const float*)d_in[1];
    const float* in_b  = (const float*)d_in[2];
    const float* out_w = (const float*)d_in[3];
    const float* out_b = (const float*)d_in[4];
    const float* f_w1  = (const float*)d_in[5];
    const float* f_b1  = (const float*)d_in[6];
    const float* f_w2  = (const float*)d_in[7];
    const float* f_b2  = (const float*)d_in[8];
    const float* f_w3  = (const float*)d_in[9];
    const float* f_b3  = (const float*)d_in[10];

    char* ws = (char*)d_ws;
    size_t o = 0;
    auto alloc = [&](size_t bytes) -> char* {
        char* p = ws + o;
        o += (bytes + 255) & ~(size_t)255;
        return p;
    };
    __bf16* xb    = (__bf16*)alloc((size_t)BL * D_MODEL * 2);
    __bf16* winb  = (__bf16*)alloc((size_t)D_MODEL * NPROJ * 2);
    __bf16* woutb = (__bf16*)alloc((size_t)D_MODEL * D_MODEL * 2);
    __bf16* hrev  = (__bf16*)alloc((size_t)2 * D_MODEL * L_SEQ * 2);
    float*  vbuf  = (float*)alloc((size_t)B_SZ * D_MODEL * L_SEQ * 4);
    __bf16* z1    = (__bf16*)alloc((size_t)B_SZ * D_MODEL * L_SEQ * 2);
    __bf16* z2    = (__bf16*)alloc((size_t)B_SZ * D_MODEL * L_SEQ * 2);
    float*  y1    = (float*)alloc((size_t)B_SZ * D_MODEL * L_SEQ * 4);
    float*  y2    = vbuf;  // v consumed by conv order-0; reuse for y2
    __bf16* yb    = z1;    // z1 consumed by conv order-0; reuse for yb

    // 1) precision converts
    f32_to_bf16_v4<<<(BL * D_MODEL / 4) / 256, 256, 0, stream>>>(
        x, xb, BL * D_MODEL / 4);
    f32_to_bf16_v4<<<(D_MODEL * NPROJ / 4) / 256, 256, 0, stream>>>(
        in_w, winb, D_MODEL * NPROJ / 4);
    f32_to_bf16_v4<<<(D_MODEL * D_MODEL / 4) / 256, 256, 0, stream>>>(
        out_w, woutb, D_MODEL * D_MODEL / 4);

    // 2) positional filter MLP -> reversed bf16 kernels
    hyena_filter_mlp<<<L_SEQ, 256, 0, stream>>>(f_w1, f_b1, f_w2, f_b2,
                                                f_w3, f_b3, hrev);

    // 3) in_proj GEMM with split/transpose epilogue
    gemm_bf16_wmma<<<dim3(NPROJ / 64, BL / 128), 256, 0, stream>>>(
        xb, winb, in_b, D_MODEL, NPROJ, 0, vbuf, z1, z2, nullptr);

    // 4) order-0 and order-1 conv + gate (Toeplitz WMMA, packed N)
    conv_gate_wmma<<<dim3(D_MODEL, 4), 256, CONV_LDS_BYTES, stream>>>(
        z1, hrev, vbuf, y1);
    conv_gate_wmma<<<dim3(D_MODEL, 4), 256, CONV_LDS_BYTES, stream>>>(
        z2, hrev + (size_t)D_MODEL * L_SEQ, y1, y2);

    // 5) layout change for out_proj
    bdl_to_bld_bf16<<<(BL * D_MODEL) / 256, 256, 0, stream>>>(y2, yb);

    // 6) out_proj GEMM -> d_out (B,L,D) fp32
    gemm_bf16_wmma<<<dim3(D_MODEL / 64, BL / 128), 256, 0, stream>>>(
        yb, woutb, out_b, D_MODEL, D_MODEL, 1, nullptr, nullptr, nullptr,
        (float*)d_out);
}
